// PredictionLayer_37993280701152
// MI455X (gfx1250) — matrix operand order repaired and checked by reading, not compiled, not run on hardware
//
#include <hip/hip_runtime.h>

// ---------------------------------------------------------------------------
// PredictionLayer for MI455X (gfx1250, wave32, WMMA + TDM).
//
// Heavy path: per 32-token tile, h = relu(x@W1+b1) via v_wmma_f32_16x16x32_bf16
// (bf16 A/B, f32 accum). LayerNorm + @W2 folded algebraically into 3 per-token
// running sums so h [32768 x 2048] is never materialized. W1 is pre-transposed
// to [N][K] bf16 so both WMMA operands stream from LDS as ds_load_b128; W1
// K-panels are DMA'd into LDS by the Tensor Data Mover (tensor_load_to_lds,
// double-buffered, TENSORcnt-tracked). 32-token tiles halve W1 L2 re-reads
// (~4 GB/head) and give 2 WMMAs per staged B fragment.
// ---------------------------------------------------------------------------

typedef __attribute__((ext_vector_type(16))) __bf16 bf16x16;
typedef __attribute__((ext_vector_type(8)))  float  f32x8;
typedef unsigned int v4u __attribute__((ext_vector_type(4)));
typedef int          v8i __attribute__((ext_vector_type(8)));
typedef int          v4i __attribute__((ext_vector_type(4)));

constexpr int Bn  = 8;
constexpr int S   = 4096;
constexpr int D   = 1024;
constexpr int H2  = 2048;
constexpr int TOK = Bn * S;          // 32768 tokens
constexpr int TM  = 32;              // tokens per workgroup
constexpr float BIGF   = 1e30f;
constexpr float LN_EPS = 1e-12f;

// B-panel LDS geometry: 128 N-rows x 32 K (bf16, K-contiguous), 64B row + 16B
// pad = 80B stride (matches TDM pad_interval=16DW, pad_amount=4DW).
constexpr int BT_ROW  = 40;                      // bf16 elements per padded row
constexpr int BT_BUFB = 128 * BT_ROW * 2;        // bytes per buffer (10240)

// ---------------------------------------------------------------------------
// Kernel 1: one-time f32 [K][N] -> bf16 [N][K] transpose of start/end W1.
// Tiled through LDS so both global sides stay coalesced. grid (32, 64).
// ---------------------------------------------------------------------------
__global__ __launch_bounds__(256)
void transpose_w1_kernel(const float* __restrict__ a, const float* __restrict__ b,
                         __bf16* __restrict__ oa, __bf16* __restrict__ ob) {
  __shared__ float t[32][33];
  const int tid = threadIdx.x;
  const int c  = tid & 31;
  const int r0 = tid >> 5;           // 0..7
  const int kb = blockIdx.x * 32;    // K tile base
  const int nb = blockIdx.y * 32;    // N tile base
#pragma unroll
  for (int p = 0; p < 4; ++p) {
    int r = r0 + p * 8;
    t[r][c] = a[(size_t)(kb + r) * H2 + nb + c];
  }
  __syncthreads();
#pragma unroll
  for (int p = 0; p < 4; ++p) {
    int rr = r0 + p * 8;
    oa[(size_t)(nb + rr) * D + kb + c] = (__bf16)t[c][rr];
  }
  __syncthreads();
#pragma unroll
  for (int p = 0; p < 4; ++p) {
    int r = r0 + p * 8;
    t[r][c] = b[(size_t)(kb + r) * H2 + nb + c];
  }
  __syncthreads();
#pragma unroll
  for (int p = 0; p < 4; ++p) {
    int rr = r0 + p * 8;
    ob[(size_t)(nb + rr) * D + kb + c] = (__bf16)t[c][rr];
  }
}

// ---------------------------------------------------------------------------
// Kernel 2: fold LN affine + W2 column:  gw2[n] = g[n]*w2[n],
// scal = [ G_s = sum(g*w2), C_s = sum(b*w2)+b2,  G_e, C_e ].
// ---------------------------------------------------------------------------
__global__ __launch_bounds__(256)
void head_prep_kernel(const float* __restrict__ gs, const float* __restrict__ bs,
                      const float* __restrict__ w2s, const float* __restrict__ b2s,
                      const float* __restrict__ ge, const float* __restrict__ be,
                      const float* __restrict__ w2e, const float* __restrict__ b2e,
                      float* __restrict__ gw2s, float* __restrict__ gw2e,
                      float* __restrict__ scal) {
  __shared__ float red[4][256];
  int tid = threadIdx.x;
  float p0 = 0.f, p1 = 0.f, p2 = 0.f, p3 = 0.f;
  for (int n = tid; n < H2; n += 256) {
    float ws = w2s[n], we = w2e[n];
    float v0 = gs[n] * ws; gw2s[n] = v0; p0 += v0; p1 += bs[n] * ws;
    float v1 = ge[n] * we; gw2e[n] = v1; p2 += v1; p3 += be[n] * we;
  }
  red[0][tid] = p0; red[1][tid] = p1; red[2][tid] = p2; red[3][tid] = p3;
  __syncthreads();
  for (int s = 128; s > 0; s >>= 1) {
    if (tid < s)
      for (int q = 0; q < 4; ++q) red[q][tid] += red[q][tid + s];
    __syncthreads();
  }
  if (tid == 0) {
    scal[0] = red[0][0];
    scal[1] = red[1][0] + b2s[0];
    scal[2] = red[2][0];
    scal[3] = red[3][0] + b2e[0];
  }
}

// ---------------------------------------------------------------------------
// Kernel 3: fused start/end head. grid = (TOK/32, 2 heads), 256 thr (8 waves).
// Fragment packing used here (A and B share contiguous k order):
//   A (16x32 bf16): lane%16 = M row; elem e -> K = e + 16*(lane>=16)
//   B (32x16 bf16): lane%16 = N col; elem e -> K = e + 16*(lane>=16)
//   C/D (16x16 f32): elem v -> row M = v + 8*(lane>=16), col N = lane%16
// W1 panels (128 N x 32 K bf16, K-contiguous) staged by TDM into padded LDS;
// each staged B fragment feeds 2 WMMAs (two 16-token row tiles).
// ---------------------------------------------------------------------------
__global__ __launch_bounds__(256)
void fused_span_head_kernel(const float* __restrict__ x,
                            const float* __restrict__ mask,
                            const __bf16* __restrict__ w1ts,
                            const __bf16* __restrict__ w1te,
                            const float* __restrict__ b1s,
                            const float* __restrict__ b1e,
                            const float* __restrict__ gw2s,
                            const float* __restrict__ gw2e,
                            const float* __restrict__ scal,
                            float* __restrict__ out) {
  constexpr int XAP = D + 8;                        // padded A stride (bf16)
  __shared__ __align__(16) __bf16 xA[TM][XAP];      // ~66 KB
  __shared__ __align__(16) __bf16 Bt[2][128][BT_ROW]; // ~20.5 KB, TDM target
  __shared__ float s1[TM], s2[TM], s3[TM];

  const int tid  = threadIdx.x;
  const int lane = tid & 31;
  const int wid  = tid >> 5;
  const int kh   = lane >> 4;        // lane half (0/1)
  const int l15  = lane & 15;
  const int head = blockIdx.y;
  const int t0   = blockIdx.x * TM;

  const __bf16* __restrict__ w1  = head ? w1te : w1ts;  // [N=2048][K=1024] bf16
  const float*  __restrict__ b1  = head ? b1e  : b1s;
  const float*  __restrict__ gw2 = head ? gw2e : gw2s;

  if (tid < TM) { s1[tid] = 0.f; s2[tid] = 0.f; s3[tid] = 0.f; }

  // Stage A tile: 32 tokens x 1024 K, f32 -> bf16 in LDS (x read once/tile).
  {
    const int tr = tid >> 3;            // 0..31 token row
    const int cb = (tid & 7) * 128;     // 128 cols per thread
    const float4* xrow =
        reinterpret_cast<const float4*>(x + (size_t)(t0 + tr) * D + cb);
#pragma unroll
    for (int i = 0; i < 32; ++i) {
      float4 f = xrow[i];
      int c = cb + i * 4;
      xA[tr][c + 0] = (__bf16)f.x; xA[tr][c + 1] = (__bf16)f.y;
      xA[tr][c + 2] = (__bf16)f.z; xA[tr][c + 3] = (__bf16)f.w;
    }
  }
  __syncthreads();

  const unsigned int bt_base =
      (unsigned int)(uintptr_t)(void*)&Bt[0][0][0];  // LDS byte offset

  for (int nstep = 0; nstep < 16; ++nstep) {
    const int n0  = nstep * 128;
    const int n0w = n0 + wid * 16;
    f32x8 acc0 = {}, acc1 = {};

    // TDM: DMA one 128(N) x 32(K) bf16 panel of transposed W1 into LDS buffer.
    // D# per ISA 8.3/8.4: count=1,type=2 | lds_addr | global_addr(57b).
    // data_size=2B; pad_enable, interval=16DW(code 3), amount=4DW(code 3);
    // tensor_dim0(K)=1024 stride 1024, tensor_dim1(N)=2048; tile 32 x 128.
    auto tdm_stage = [&](int k0, int buf) {
      unsigned long long gaddr = (unsigned long long)(uintptr_t)w1 +
          ((unsigned long long)n0 * (unsigned int)D + (unsigned int)k0) * 2ull;
      unsigned int lds = bt_base + (unsigned int)buf * (unsigned int)BT_BUFB;
      v4u g0;
      g0[0] = 1u;                                       // count=1 (valid D#)
      g0[1] = lds;                                      // lds_addr
      g0[2] = (unsigned int)gaddr;                      // global_addr[31:0]
      g0[3] = (unsigned int)(gaddr >> 32) | (2u << 30); // ga[56:32] | type=2
      v8i g1;
      g1[0] = (int)((1u << 16) | (1u << 20) | (3u << 22) | (3u << 25));
      g1[1] = (int)(1024u << 16);                       // tensor_dim0 lo16
      g1[2] = (int)(2048u << 16);                       // td0 hi=0 | td1 lo16
      g1[3] = (int)(32u << 16);                         // td1 hi=0 | tile_dim0
      g1[4] = 128;                                      // tile_dim1 (dim2=0)
      g1[5] = 1024;                                     // tensor_dim0_stride
      g1[6] = 0;
      g1[7] = 0;
      v4i gz = {0, 0, 0, 0};
      asm volatile("tensor_load_to_lds %0, %1, %2, %3"
                   :: "s"(g0), "s"(g1), "s"(gz), "s"(gz)
                   : "memory");
    };

    if (wid == 0) {
      tdm_stage(0, 0);
      __builtin_amdgcn_s_wait_tensorcnt(0);
    }
    __syncthreads();

    for (int ks = 0; ks < 32; ++ks) {
      const int cur = ks & 1;
      if (wid == 0 && ks + 1 < 32) tdm_stage((ks + 1) * 32, cur ^ 1);

      const int k0 = ks * 32;
      bf16x16 a0, a1, b;
#pragma unroll
      for (int e = 0; e < 16; ++e) {                    // 2x ds_load_b128 each
        int kl = e + (kh ? 16 : 0);
        a0[e] = xA[l15][k0 + kl];
        a1[e] = xA[16 + l15][k0 + kl];
      }
      const __bf16* bp = &Bt[cur][wid * 16 + l15][kh * 16];
#pragma unroll
      for (int e = 0; e < 16; ++e) b[e] = bp[e];        // 2x ds_load_b128
      acc0 = __builtin_amdgcn_wmma_f32_16x16x32_bf16(
          false, a0, false, b, (short)0, acc0, false, false);
      acc1 = __builtin_amdgcn_wmma_f32_16x16x32_bf16(
          false, a1, false, b, (short)0, acc1, false, false);

      if (wid == 0 && ks + 1 < 32) __builtin_amdgcn_s_wait_tensorcnt(0);
      __syncthreads();
    }

    // Epilogue: bias + relu, fold both row tiles into per-token running sums.
    const float bias = b1[n0w + l15];
    const float gwv  = gw2[n0w + l15];
#pragma unroll
    for (int t = 0; t < 2; ++t) {
      const f32x8& acc = t ? acc1 : acc0;
#pragma unroll
      for (int v = 0; v < 8; ++v) {
        float hv = fmaxf(acc[v] + bias, 0.f);
        float t1 = hv, t2 = hv * hv, t3 = hv * gwv;
#pragma unroll
        for (int m = 8; m >= 1; m >>= 1) {              // reduce 16 N-cols
          t1 += __shfl_xor(t1, m, 32);
          t2 += __shfl_xor(t2, m, 32);
          t3 += __shfl_xor(t3, m, 32);
        }
        if (l15 == 0) {
          int row = t * 16 + v + (kh ? 8 : 0);
          atomicAdd(&s1[row], t1);
          atomicAdd(&s2[row], t2);
          atomicAdd(&s3[row], t3);
        }
      }
    }
  }

  __syncthreads();
  if (tid < TM) {
    const int token = t0 + tid;
    const float G = scal[head * 2 + 0];
    const float C = scal[head * 2 + 1];
    float mu   = s1[tid] * (1.0f / (float)H2);
    float var  = s2[tid] * (1.0f / (float)H2) - mu * mu;
    float rstd = rsqrtf(var + LN_EPS);
    float val  = rstd * (s3[tid] - mu * G) + C;
    val -= BIGF * (1.0f - mask[token]);
    out[head * TOK + token] = val;
  }
}

// ---------------------------------------------------------------------------
// Kernel 4: type head on x[:,0,:] (8 rows). <0.01% of FLOPs -> scalar f32.
// ---------------------------------------------------------------------------
__global__ __launch_bounds__(256)
void type_head_kernel(const float* __restrict__ x,
                      const float* __restrict__ w1, const float* __restrict__ b1,
                      const float* __restrict__ g,  const float* __restrict__ bsh,
                      const float* __restrict__ w2, const float* __restrict__ b2,
                      float* __restrict__ out) {
  __shared__ float xr[D];
  __shared__ float h[H2];
  __shared__ float red[256];
  __shared__ float stat[2];
  const int tid = threadIdx.x;

  for (int r = 0; r < Bn; ++r) {
    for (int k = tid; k < D; k += 256) xr[k] = x[(size_t)r * S * D + k];
    __syncthreads();
    for (int col = tid; col < H2; col += 256) {
      float a = b1[col];
      for (int k = 0; k < D; ++k) a = fmaf(xr[k], w1[(size_t)k * H2 + col], a);
      h[col] = fmaxf(a, 0.f);
    }
    __syncthreads();
    float p1 = 0.f, p2 = 0.f;
    for (int c = tid; c < H2; c += 256) { float v = h[c]; p1 += v; p2 += v * v; }
    red[tid] = p1; __syncthreads();
    for (int s = 128; s > 0; s >>= 1) { if (tid < s) red[tid] += red[tid + s]; __syncthreads(); }
    if (tid == 0) stat[0] = red[0];
    __syncthreads();
    red[tid] = p2; __syncthreads();
    for (int s = 128; s > 0; s >>= 1) { if (tid < s) red[tid] += red[tid + s]; __syncthreads(); }
    if (tid == 0) stat[1] = red[0];
    __syncthreads();
    float mu   = stat[0] / (float)H2;
    float var  = stat[1] / (float)H2 - mu * mu;
    float rstd = rsqrtf(var + LN_EPS);
    float p[4] = {0.f, 0.f, 0.f, 0.f};
    for (int c = tid; c < H2; c += 256) {
      float hn = (h[c] - mu) * rstd * g[c] + bsh[c];
#pragma unroll
      for (int a2 = 0; a2 < 4; ++a2) p[a2] += hn * w2[c * 4 + a2];
    }
    for (int a2 = 0; a2 < 4; ++a2) {
      red[tid] = p[a2]; __syncthreads();
      for (int s = 128; s > 0; s >>= 1) { if (tid < s) red[tid] += red[tid + s]; __syncthreads(); }
      if (tid == 0) out[r * 4 + a2] = red[0] + b2[a2];
      __syncthreads();
    }
  }
}

// ---------------------------------------------------------------------------
// Kernel 5: banded argmax; reproduces jnp.argmax tie-breaking via global band
// max M, then min-index matches. Indices written as floats.
// ---------------------------------------------------------------------------
__global__ __launch_bounds__(256)
void band_argmax_kernel(float* __restrict__ out) {
  const int b   = blockIdx.x;
  const int tid = threadIdx.x;
  const float* sp = out + (size_t)b * S;
  const float* ep = out + (size_t)TOK + (size_t)b * S;
  __shared__ float red[256];
  __shared__ int bi, bj;

  float m = -3.4e38f;
  for (int i = tid; i < S; i += 256) {
    float em = -3.4e38f;
    int jend = min(i + 15, S - 1);
    for (int j = i; j <= jend; ++j) em = fmaxf(em, ep[j]);
    m = fmaxf(m, sp[i] + em);
  }
  red[tid] = m; __syncthreads();
  for (int s = 128; s > 0; s >>= 1) { if (tid < s) red[tid] = fmaxf(red[tid], red[tid + s]); __syncthreads(); }
  const float M = red[0];
  if (tid == 0) { bi = 0x7fffffff; bj = 0x7fffffff; }
  __syncthreads();

  for (int i = tid; i < S; i += 256) {
    float em = -3.4e38f;
    int jend = min(i + 15, S - 1);
    for (int j = i; j <= jend; ++j) em = fmaxf(em, ep[j]);
    if (sp[i] + em == M) atomicMin(&bi, i);
  }
  for (int j = tid; j < S; j += 256) {
    float sm = -3.4e38f;
    int i0 = max(j - 15, 0);
    for (int i = i0; i <= j; ++i) sm = fmaxf(sm, sp[i]);
    if (ep[j] + sm == M) atomicMin(&bj, j);
  }
  __syncthreads();
  if (tid == 0) {
    out[2 * TOK + 32 + b]     = (float)bi;  // yp1
    out[2 * TOK + 32 + 8 + b] = (float)bj;  // yp2
  }
}

// ---------------------------------------------------------------------------
// d_out layout (float, 65584): start[32768] | end[32768] | type[32] |
// yp1[8] | yp2[8].  Workspace: W1s^T bf16 (4MB) | W1e^T bf16 (4MB) |
// gw2_s[2048] | gw2_e[2048] | 4 scalars.
// ---------------------------------------------------------------------------
extern "C" void kernel_launch(void* const* d_in, const int* in_sizes, int n_in,
                              void* d_out, int out_size, void* d_ws, size_t ws_size,
                              hipStream_t stream) {
  const float* x    = (const float*)d_in[0];
  const float* mask = (const float*)d_in[1];
  const float* sw1  = (const float*)d_in[2];
  const float* sb1  = (const float*)d_in[3];
  const float* sg   = (const float*)d_in[4];
  const float* sb   = (const float*)d_in[5];
  const float* sw2  = (const float*)d_in[6];
  const float* sb2  = (const float*)d_in[7];
  const float* ew1  = (const float*)d_in[8];
  const float* eb1  = (const float*)d_in[9];
  const float* eg   = (const float*)d_in[10];
  const float* eb   = (const float*)d_in[11];
  const float* ew2  = (const float*)d_in[12];
  const float* eb2  = (const float*)d_in[13];
  const float* tw1  = (const float*)d_in[14];
  const float* tb1  = (const float*)d_in[15];
  const float* tg   = (const float*)d_in[16];
  const float* tb   = (const float*)d_in[17];
  const float* tw2  = (const float*)d_in[18];
  const float* tb2  = (const float*)d_in[19];

  float* out = (float*)d_out;

  const size_t W1N = (size_t)D * H2;
  __bf16* w1s_h = (__bf16*)d_ws;
  __bf16* w1e_h = w1s_h + W1N;
  float*  gw2s  = (float*)(w1e_h + W1N);
  float*  gw2e  = gw2s + H2;
  float*  scal  = gw2e + H2;

  transpose_w1_kernel<<<dim3(32, 64), 256, 0, stream>>>(sw1, ew1, w1s_h, w1e_h);
  head_prep_kernel<<<1, 256, 0, stream>>>(sg, sb, sw2, sb2, eg, eb, ew2, eb2,
                                          gw2s, gw2e, scal);
  fused_span_head_kernel<<<dim3(TOK / TM, 2), 256, 0, stream>>>(
      x, mask, w1s_h, w1e_h, sb1, eb1, gw2s, gw2e, scal, out);
  type_head_kernel<<<1, 256, 0, stream>>>(x, tw1, tb1, tg, tb, tw2, tb2,
                                          out + 2 * TOK);
  band_argmax_kernel<<<8, 256, 0, stream>>>(out);
}
